// Block_4277787427205
// MI455X (gfx1250) — compile-verified, matrix operand order checked by reading, not run on hardware
//
#include <hip/hip_runtime.h>
#include <hip/hip_bf16.h>
#include <math.h>

typedef __attribute__((ext_vector_type(16))) _Float16 v16h;
typedef __attribute__((ext_vector_type(8)))  _Float16 v8h;
typedef __attribute__((ext_vector_type(8)))  float    v8f;

union FragU { v16h v; v8h h2[2]; };

// Load a 16-bit A/B WMMA fragment per the CDNA5 ISA layout:
// lane (h = lane>>4, r = lane&15) holds row/col r, K chunks {koff..koff+7} in
// VGPRs 0-3 and {koff+16..koff+23} in VGPRs 4-7, where koff = kbase + 8*h.
__device__ __forceinline__ v16h load_frag16(const _Float16* rowp, int koff) {
    FragU f;
    f.h2[0] = *(const v8h*)(rowp + koff);
    f.h2[1] = *(const v8h*)(rowp + koff + 16);
    return f.v;
}

__device__ __forceinline__ v8f wmma_f16(v16h a, v16h b, v8f c) {
    return __builtin_amdgcn_wmma_f32_16x16x32_f16(false, a, false, b,
                                                  (short)0, c, false, false);
}

enum { EPI_F16 = 0, EPI_F16_GELU = 1, EPI_F32_RES = 2 };

template<int EPI>
__device__ __forceinline__ void store_tile(v8f c, int row0, int col, int N,
        const float* __restrict__ bias, const float* __restrict__ res,
        _Float16* __restrict__ out16, float* __restrict__ out32) {
    float bcol = bias[col];
    #pragma unroll
    for (int i = 0; i < 8; ++i) {
        size_t idx = (size_t)(row0 + i) * N + col;
        float acc = c[i] + bcol;
        if (EPI == EPI_F16) {
            out16[idx] = (_Float16)acc;
        } else if (EPI == EPI_F16_GELU) {
            float g = 0.5f * acc * (1.0f + erff(acc * 0.70710678118f));
            out16[idx] = (_Float16)g;
        } else {
            out32[idx] = acc + res[idx];
        }
    }
}

// C = A(f16, MxK row-major) @ W^T (W: f16, NxK row-major) + bias [+res]
// One wave per 32x64 output tile: 2 A frags + 4 B frags -> 8 WMMAs per K=32
// step (0.75 fragment loads per WMMA).
template<int EPI>
__global__ __launch_bounds__(256)
void gemm_wmma_kernel(const _Float16* __restrict__ A, const _Float16* __restrict__ W,
                      const float* __restrict__ bias, const float* __restrict__ res,
                      _Float16* __restrict__ out16, float* __restrict__ out32,
                      int M, int N, int K) {
    const int lane  = threadIdx.x & 31;
    const int wave  = blockIdx.x * 8 + (threadIdx.x >> 5);
    const int tilesN = N >> 6;
    const int total  = (M >> 5) * tilesN;
    if (wave >= total) return;              // uniform per wave -> EXEC stays all-1s
    const int mt = wave / tilesN;
    const int nt = wave % tilesN;
    const int h  = lane >> 4;
    const int r  = lane & 15;

    const _Float16* arow0 = A + (size_t)(mt * 32 + r) * K;
    const _Float16* arow1 = A + (size_t)(mt * 32 + 16 + r) * K;
    const _Float16* wrow  = W + (size_t)(nt * 64 + r) * K;

    v8f c00 = {}, c01 = {}, c02 = {}, c03 = {};
    v8f c10 = {}, c11 = {}, c12 = {}, c13 = {};
    for (int k0 = 0; k0 < K; k0 += 32) {
        if (k0 + 128 < K) {
            __builtin_prefetch(arow0 + k0 + 128, 0, 0);  // global_prefetch_b8
            __builtin_prefetch(arow1 + k0 + 128, 0, 0);
            __builtin_prefetch(wrow  + k0 + 128, 0, 0);
        }
        const int ko = k0 + 8 * h;
        v16h a0 = load_frag16(arow0, ko);
        v16h a1 = load_frag16(arow1, ko);
        v16h b0 = load_frag16(wrow,                  ko);
        v16h b1 = load_frag16(wrow + (size_t)16 * K, ko);
        v16h b2 = load_frag16(wrow + (size_t)32 * K, ko);
        v16h b3 = load_frag16(wrow + (size_t)48 * K, ko);
        c00 = wmma_f16(a0, b0, c00);  c10 = wmma_f16(a1, b0, c10);
        c01 = wmma_f16(a0, b1, c01);  c11 = wmma_f16(a1, b1, c11);
        c02 = wmma_f16(a0, b2, c02);  c12 = wmma_f16(a1, b2, c12);
        c03 = wmma_f16(a0, b3, c03);  c13 = wmma_f16(a1, b3, c13);
    }
    const int row0 = mt * 32 + 8 * h;       // C/D layout: VGPR i -> row i (+8 hi half)
    const int row1 = row0 + 16;
    const int colb = nt * 64 + r;
    store_tile<EPI>(c00, row0, colb +  0, N, bias, res, out16, out32);
    store_tile<EPI>(c01, row0, colb + 16, N, bias, res, out16, out32);
    store_tile<EPI>(c02, row0, colb + 32, N, bias, res, out16, out32);
    store_tile<EPI>(c03, row0, colb + 48, N, bias, res, out16, out32);
    store_tile<EPI>(c10, row1, colb +  0, N, bias, res, out16, out32);
    store_tile<EPI>(c11, row1, colb + 16, N, bias, res, out16, out32);
    store_tile<EPI>(c12, row1, colb + 32, N, bias, res, out16, out32);
    store_tile<EPI>(c13, row1, colb + 48, N, bias, res, out16, out32);
}

// Flash attention: 4 waves per block, one (batch, head, 16-query tile) per wave.
// qkv: [4096, 2304] f16 (q|k|v per token). vT: per (b,h) 64x2048 f16.
// Row max is shuffle-reduced per step; row sum is kept as per-lane partials and
// reduced once at the end (halves ds_bpermute traffic in the hot loop).
__global__ __launch_bounds__(128)
void attn_kernel(const _Float16* __restrict__ qkv, const _Float16* __restrict__ vT,
                 _Float16* __restrict__ o) {
    __shared__ __align__(16) _Float16 plds[4][16 * 32];
    const int widx = threadIdx.x >> 5;
    const int lane = threadIdx.x & 31;
    const int h = lane >> 4, r = lane & 15;
    const int gw   = blockIdx.x * 4 + widx;  // 0..3071
    const int qt   = gw & 127;               // 2048/16 query tiles
    const int bh   = gw >> 7;                // 0..23
    const int head = bh % 12;
    const int b    = bh / 12;
    const int LDQ  = 2304;
    _Float16* pbuf = plds[widx];

    const _Float16* qbase = qkv + (size_t)(b * 2048 + qt * 16) * LDQ + head * 64;
    const _Float16* kbase = qkv + (size_t)(b * 2048) * LDQ + 768 + head * 64;
    const _Float16* vbase = vT  + (size_t)bh * 64 * 2048;

    // Q fragments (16 rows x 64 d) are loop-invariant: preload both K=32 slices.
    const _Float16* qrow = qbase + (size_t)r * LDQ;
    v16h qa0 = load_frag16(qrow,  0 + 8 * h);
    v16h qa1 = load_frag16(qrow, 32 + 8 * h);

    v8f o0 = {}, o1 = {}, o2 = {}, o3 = {};
    float mrun[8], lpart[8];
    #pragma unroll
    for (int i = 0; i < 8; ++i) { mrun[i] = -1e30f; lpart[i] = 0.0f; }
    const float scale = 0.125f;              // 64^-0.5

    for (int j = 0; j < 2048; j += 32) {
        // S tile 16x32 = two 16x16 D tiles, K(=HEAD_DIM) split 2x32.
        v8f s0 = {}, s1 = {};
        {
            const _Float16* k0 = kbase + (size_t)(j + r) * LDQ;
            const _Float16* k1 = kbase + (size_t)(j + 16 + r) * LDQ;
            v16h kb00 = load_frag16(k0,  0 + 8 * h);
            v16h kb01 = load_frag16(k0, 32 + 8 * h);
            v16h kb10 = load_frag16(k1,  0 + 8 * h);
            v16h kb11 = load_frag16(k1, 32 + 8 * h);
            s0 = wmma_f16(qa0, kb00, s0);  s0 = wmma_f16(qa1, kb01, s0);
            s1 = wmma_f16(qa0, kb10, s1);  s1 = wmma_f16(qa1, kb11, s1);
        }
        // Online softmax: each row lives across the 16 lanes of one half-wave.
        float alpha[8], p0[8], p1[8];
        #pragma unroll
        for (int i = 0; i < 8; ++i) {
            float a0 = s0[i] * scale, a1 = s1[i] * scale;
            float mx = fmaxf(a0, a1);
            #pragma unroll
            for (int mk = 1; mk < 16; mk <<= 1) mx = fmaxf(mx, __shfl_xor(mx, mk, 32));
            float mnew = fmaxf(mrun[i], mx);
            float e0 = __expf(a0 - mnew);
            float e1 = __expf(a1 - mnew);
            alpha[i] = __expf(mrun[i] - mnew);
            lpart[i] = lpart[i] * alpha[i] + (e0 + e1);  // per-lane partial sum
            mrun[i]  = mnew;
            p0[i] = e0; p1[i] = e1;
        }
        #pragma unroll
        for (int i = 0; i < 8; ++i) {
            o0[i] *= alpha[i]; o1[i] *= alpha[i];
            o2[i] *= alpha[i]; o3[i] *= alpha[i];
        }
        // Re-layout P: D layout -> row-major [16][32] in LDS -> A fragment.
        __builtin_amdgcn_wave_barrier();
        #pragma unroll
        for (int i = 0; i < 8; ++i) {
            pbuf[(i + 8 * h) * 32 + r]      = (_Float16)p0[i];
            pbuf[(i + 8 * h) * 32 + 16 + r] = (_Float16)p1[i];
        }
        __builtin_amdgcn_wave_barrier();
        v16h pa = load_frag16(&pbuf[r * 32], 8 * h);
        __builtin_amdgcn_wave_barrier();
        // O += P(16x32 keys) @ V(32 keys x 64 d); vT rows contiguous in key.
        #pragma unroll
        for (int t = 0; t < 4; ++t) {
            const _Float16* vrow = vbase + (size_t)(t * 16 + r) * 2048;
            v16h vb = load_frag16(vrow, j + 8 * h);
            v8f& ot = (t == 0 ? o0 : (t == 1 ? o1 : (t == 2 ? o2 : o3)));
            ot = wmma_f16(pa, vb, ot);
        }
    }
    #pragma unroll
    for (int i = 0; i < 8; ++i) {
        float l = lpart[i];                   // final cross-lane row-sum reduction
        #pragma unroll
        for (int mk = 1; mk < 16; mk <<= 1) l += __shfl_xor(l, mk, 32);
        float inv = 1.0f / l;
        int token = b * 2048 + qt * 16 + i + 8 * h;
        size_t base = (size_t)token * 768 + head * 64;
        o[base +  0 + r] = (_Float16)(o0[i] * inv);
        o[base + 16 + r] = (_Float16)(o1[i] * inv);
        o[base + 32 + r] = (_Float16)(o2[i] * inv);
        o[base + 48 + r] = (_Float16)(o3[i] * inv);
    }
}

__global__ __launch_bounds__(256)
void ln_kernel(const float* __restrict__ x, const float* __restrict__ g,
               const float* __restrict__ bta, _Float16* __restrict__ out) {
    const int C = 768;
    const int row = blockIdx.x;
    const float* xr = x + (size_t)row * C;
    float s = 0.f, s2 = 0.f;
    for (int c = threadIdx.x; c < C; c += 256) { float v = xr[c]; s += v; s2 += v * v; }
    __shared__ float r1[256], r2[256];
    r1[threadIdx.x] = s; r2[threadIdx.x] = s2;
    __syncthreads();
    for (int st = 128; st > 0; st >>= 1) {
        if ((int)threadIdx.x < st) {
            r1[threadIdx.x] += r1[threadIdx.x + st];
            r2[threadIdx.x] += r2[threadIdx.x + st];
        }
        __syncthreads();
    }
    float mu   = r1[0] * (1.0f / C);
    float var  = r2[0] * (1.0f / C) - mu * mu;
    float rstd = rsqrtf(var + 1e-5f);
    for (int c = threadIdx.x; c < C; c += 256)
        out[(size_t)row * C + c] = (_Float16)((xr[c] - mu) * rstd * g[c] + bta[c]);
}

__global__ void f32_to_f16_kernel(const float* __restrict__ src,
                                  _Float16* __restrict__ dst, int n) {
    int i = blockIdx.x * 256 + threadIdx.x;
    if (i < n) dst[i] = (_Float16)src[i];
}

// vT[((b*12+h)*64 + d)*2048 + tok] = V(token,head,d) from qkv buffer.
__global__ void transpose_v_kernel(const _Float16* __restrict__ qkv,
                                   _Float16* __restrict__ vT) {
    int idx = blockIdx.x * 256 + threadIdx.x;
    if (idx >= 2 * 12 * 64 * 2048) return;
    int tok  = idx & 2047;
    int t1   = idx >> 11;
    int d    = t1 & 63;
    int bh   = t1 >> 6;
    int head = bh % 12;
    int b    = bh / 12;
    vT[idx] = qkv[(size_t)(b * 2048 + tok) * 2304 + 1536 + head * 64 + d];
}

extern "C" void kernel_launch(void* const* d_in, const int* in_sizes, int n_in,
                              void* d_out, int out_size, void* d_ws, size_t ws_size,
                              hipStream_t stream) {
    const float* x      = (const float*)d_in[0];
    const float* ln1_g  = (const float*)d_in[1];
    const float* ln1_b  = (const float*)d_in[2];
    const float* qkv_w  = (const float*)d_in[3];
    const float* qkv_b  = (const float*)d_in[4];
    const float* proj_w = (const float*)d_in[5];
    const float* proj_b = (const float*)d_in[6];
    const float* ln2_g  = (const float*)d_in[7];
    const float* ln2_b  = (const float*)d_in[8];
    const float* fc1_w  = (const float*)d_in[9];
    const float* fc1_b  = (const float*)d_in[10];
    const float* fc2_w  = (const float*)d_in[11];
    const float* fc2_b  = (const float*)d_in[12];

    const int M = 2 * 2048, E = 768, QKV = 2304, HID = 3072;

    char* ws = (char*)d_ws;
    size_t off = 0;
    auto take = [&](size_t bytes) -> char* {
        char* p = ws + off;
        off = (off + bytes + 255) & ~(size_t)255;
        return p;
    };
    _Float16* wqkv_h  = (_Float16*)take((size_t)QKV * E * 2);
    _Float16* wproj_h = (_Float16*)take((size_t)E * E * 2);
    _Float16* wfc1_h  = (_Float16*)take((size_t)HID * E * 2);
    _Float16* wfc2_h  = (_Float16*)take((size_t)E * HID * 2);
    _Float16* h1_h    = (_Float16*)take((size_t)M * E * 2);
    _Float16* qkv_h   = (_Float16*)take((size_t)M * QKV * 2);
    _Float16* vT_h    = (_Float16*)take((size_t)M * E * 2);
    _Float16* o_h     = (_Float16*)take((size_t)M * E * 2);
    float*    x1_f    = (float*)   take((size_t)M * E * 4);
    _Float16* h2_h    = (_Float16*)take((size_t)M * E * 2);
    _Float16* act_h   = (_Float16*)take((size_t)M * HID * 2);

    auto cvt = [&](const float* s, _Float16* d, int n) {
        f32_to_f16_kernel<<<(n + 255) / 256, 256, 0, stream>>>(s, d, n);
    };
    cvt(qkv_w,  wqkv_h,  QKV * E);
    cvt(proj_w, wproj_h, E * E);
    cvt(fc1_w,  wfc1_h,  HID * E);
    cvt(fc2_w,  wfc2_h,  E * HID);

    ln_kernel<<<M, 256, 0, stream>>>(x, ln1_g, ln1_b, h1_h);

    auto gblocks = [](int m, int n) { return ((m / 32) * (n / 64) + 7) / 8; };

    gemm_wmma_kernel<EPI_F16><<<gblocks(M, QKV), 256, 0, stream>>>(
        h1_h, wqkv_h, qkv_b, nullptr, qkv_h, nullptr, M, QKV, E);

    transpose_v_kernel<<<(M * E + 255) / 256, 256, 0, stream>>>(qkv_h, vT_h);

    attn_kernel<<<(24 * 128) / 4, 128, 0, stream>>>(qkv_h, vT_h, o_h);

    gemm_wmma_kernel<EPI_F32_RES><<<gblocks(M, E), 256, 0, stream>>>(
        o_h, wproj_h, proj_b, x, nullptr, x1_f, M, E, E);

    ln_kernel<<<M, 256, 0, stream>>>(x1_f, ln2_g, ln2_b, h2_h);

    gemm_wmma_kernel<EPI_F16_GELU><<<gblocks(M, HID), 256, 0, stream>>>(
        h2_h, wfc1_h, fc1_b, nullptr, act_h, nullptr, M, HID, E);

    gemm_wmma_kernel<EPI_F32_RES><<<gblocks(M, E), 256, 0, stream>>>(
        act_h, wfc2_h, fc2_b, x1_f, nullptr, (float*)d_out, M, E, HID);
}
